// RCCAModule_83004537963194
// MI455X (gfx1250) — matrix operand order repaired and checked by reading, not compiled, not run on hardware
//
#include <hip/hip_runtime.h>
#include <hip/hip_bf16.h>

typedef __bf16 bf16;
typedef __attribute__((ext_vector_type(16))) __bf16 v16bf;
typedef __attribute__((ext_vector_type(8)))  __bf16 v8bf;
typedef __attribute__((ext_vector_type(8)))  float  v8f;

#define BB   8
#define CC   256
#define HH   128
#define WW   128
#define CI   64
#define CQ   8
#define CO   32
#define HW   (HH*WW)
#define NPIX ((size_t)BB*HW)

// ---------------------------------------------------------------------------
// Weight packing: fp32 OIHW -> bf16 B-fragments in WMMA 16x16x32 lane layout.
// Fragment f = ((tile*taps + tap)*KC + kc); 512 bf16 per fragment.
// Lane l (col n = l&15), element e: K-in-chunk = ((l&16)?8:0) + (e<8 ? e : e+8).
// ---------------------------------------------------------------------------
__global__ void pack_conv_weights(const float* __restrict__ src, bf16* __restrict__ dst,
                                  int Cout, int Cin, int taps) {
    const int KC = Cin >> 5;
    const int total = (Cout >> 4) * taps * KC * 512;
    int idx = blockIdx.x * blockDim.x + threadIdx.x;
    if (idx >= total) return;
    const int e    = idx & 15;
    const int l    = (idx >> 4) & 31;
    const int frag = idx >> 9;
    const int kc   = frag % KC;
    const int tap  = (frag / KC) % taps;
    const int ot   = frag / (KC * taps);
    const int oc   = ot * 16 + (l & 15);
    const int cin  = kc * 32 + ((l & 16) ? 8 : 0) + ((e < 8) ? e : e + 8);
    dst[idx] = (bf16)src[((size_t)oc * Cin + cin) * taps + tap];
}

// q/k/v virtual Cout = 80 (rows 0-7 Wq, 8-15 Wk, 16-79 Wv), Cin=64 -> KC=2, 5 tiles
__global__ void pack_qkv_weights(const float* __restrict__ Wq, const float* __restrict__ Wk,
                                 const float* __restrict__ Wv, bf16* __restrict__ dst) {
    const int total = 5 * 2 * 512;
    int idx = blockIdx.x * blockDim.x + threadIdx.x;
    if (idx >= total) return;
    const int e    = idx & 15;
    const int l    = (idx >> 4) & 31;
    const int frag = idx >> 9;
    const int kc   = frag & 1;
    const int ot   = frag >> 1;
    const int oc   = ot * 16 + (l & 15);
    const int cin  = kc * 32 + ((l & 16) ? 8 : 0) + ((e < 8) ? e : e + 8);
    float v;
    if      (oc < 8)  v = Wq[oc * CI + cin];
    else if (oc < 16) v = Wk[(oc - 8) * CI + cin];
    else              v = Wv[(oc - 16) * CI + cin];
    dst[idx] = (bf16)v;
}

// ---------------------------------------------------------------------------
// conv3x3 (pad 1) implicit-GEMM. One wave = 16 pixels x (NT*16) out-channels.
// Branch-free clamped A loads (single VMEM clause) + next-chunk prefetch.
// Fuses per-channel sum / sumsq accumulation for the following ABN.
// ---------------------------------------------------------------------------
template<int NT>
__global__ void __launch_bounds__(32)
conv3x3_wmma(const float* __restrict__ in1, int Cin1,
             const float* __restrict__ in2, int Cin2,
             const bf16* __restrict__ wf, float* __restrict__ out,
             float* __restrict__ sum, float* __restrict__ sumsq, int Cout) {
    const int Cin = Cin1 + Cin2;
    const int KC  = Cin >> 5;
    const int nwt = WW / 16;
    const int wt  = blockIdx.x % nwt;
    const int og  = blockIdx.x / nwt;       // group of NT col-tiles
    const int h   = blockIdx.y;
    const int b   = blockIdx.z;
    const int lane  = threadIdx.x;
    const int w0    = wt * 16;
    const int row   = lane & 15;
    const int kbase = (lane & 16) ? 8 : 0;

    v8f acc[NT];
#pragma unroll
    for (int j = 0; j < NT; ++j) acc[j] = (v8f){};

    for (int t = 0; t < 9; ++t) {
        const int r = t / 3 - 1, s = t % 3 - 1;
        const int hr = h + r;
        if (hr < 0 || hr >= HH) continue;
        const int  wp  = w0 + row + s;
        const bool wok = (unsigned)wp < (unsigned)WW;
        const int  wpc = wok ? wp : 0;           // clamped: load always, zero later
        for (int kc = 0; kc < KC; ++kc) {
            const int c0 = kc << 5;
            const float* srcRow;                 // uniform select (Cin1 % 32 == 0)
            if (c0 < Cin1)
                srcRow = in1 + (((size_t)b * Cin1 + c0 + kbase) * HH + hr) * WW + wpc;
            else
                srcRow = in2 + (((size_t)b * Cin2 + (c0 - Cin1) + kbase) * HH + hr) * WW + wpc;
            if (kc + 1 < KC && c0 + 32 < Cin1)   // L2 prefetch of next K-chunk row
                __builtin_prefetch(srcRow + (size_t)32 * HW, 0, 1);
            v16bf a;
#pragma unroll
            for (int e = 0; e < 16; ++e) {
                const int ko = (e < 8) ? e : e + 8;      // K offset above kbase
                const float val = srcRow[(size_t)ko * HW];
                a[e] = (bf16)(wok ? val : 0.f);
            }
#pragma unroll
            for (int j = 0; j < NT; ++j) {
                const int tile = og * NT + j;
                const v16bf bfrag =
                    *(const v16bf*)(wf + ((size_t)((tile * 9 + t) * KC + kc) << 9) + (lane << 4));
                acc[j] = __builtin_amdgcn_wmma_f32_16x16x32_bf16(false, a, false, bfrag,
                                                                 (short)0, acc[j], false, false);
            }
        }
    }
    const int n     = lane & 15;
    const int mbase = (lane & 16) ? 8 : 0;
#pragma unroll
    for (int j = 0; j < NT; ++j) {
        const int oc = (og * NT + j) * 16 + n;
        float* op = out + (((size_t)b * Cout + oc) * HH + h) * WW + w0 + mbase;
        float s1 = 0.f, s2 = 0.f;
#pragma unroll
        for (int i = 0; i < 8; ++i) { s1 += acc[j][i]; s2 += acc[j][i] * acc[j][i]; }
        *(float4*)op       = make_float4(acc[j][0], acc[j][1], acc[j][2], acc[j][3]);
        *(float4*)(op + 4) = make_float4(acc[j][4], acc[j][5], acc[j][6], acc[j][7]);
        s1 += __shfl_xor(s1, 16, 32);
        s2 += __shfl_xor(s2, 16, 32);
        if (!(lane & 16)) {
            atomicAdd(&sum[oc], s1);
            atomicAdd(&sumsq[oc], s2);
        }
    }
}

// ABN normalize + leaky ReLU, float4 elementwise over (B,C,H,W) buffer
__global__ void abn_apply(float* __restrict__ buf, const float* __restrict__ sum,
                          const float* __restrict__ sumsq, const float* __restrict__ g,
                          const float* __restrict__ bta, int Cch, int total4) {
    int idx = blockIdx.x * blockDim.x + threadIdx.x;
    if (idx >= total4) return;
    const int   base = idx * 4;
    const int   c    = (base / HW) % Cch;
    const float cnt  = (float)(BB * HW);
    const float m    = sum[c] / cnt;
    const float var  = sumsq[c] / cnt - m * m;
    const float sc   = rsqrtf(var + 1e-5f) * g[c];
    const float sh   = bta[c];
    float4 t = *(float4*)(buf + base);
    float y0 = (t.x - m) * sc + sh; t.x = y0 >= 0.f ? y0 : 0.01f * y0;
    float y1 = (t.y - m) * sc + sh; t.y = y1 >= 0.f ? y1 : 0.01f * y1;
    float y2 = (t.z - m) * sc + sh; t.z = y2 >= 0.f ? y2 : 0.01f * y2;
    float y3 = (t.w - m) * sc + sh; t.w = y3 >= 0.f ? y3 : 0.01f * y3;
    *(float4*)(buf + base) = t;
}

// ---------------------------------------------------------------------------
// q/k/v pointwise GEMM via WMMA: one wave = 16 pixels x all 80 virtual outs
// (5 col-tiles share each A fragment), K = 64. Channel-last outputs.
// ---------------------------------------------------------------------------
__global__ void __launch_bounds__(32)
qkv_wmma(const float* __restrict__ x, const bf16* __restrict__ wf,
         float* __restrict__ q, float* __restrict__ k, float* __restrict__ v) {
    const int wt = blockIdx.x;                  // 0..7
    const int h  = blockIdx.y, b = blockIdx.z;
    const int lane  = threadIdx.x;
    const int w0    = wt * 16;
    const int row   = lane & 15;
    const int kbase = (lane & 16) ? 8 : 0;
    v8f acc[5];
#pragma unroll
    for (int j = 0; j < 5; ++j) acc[j] = (v8f){};
#pragma unroll
    for (int kc = 0; kc < 2; ++kc) {
        const float* srcRow = x + (((size_t)b * CI + (kc << 5) + kbase) * HH + h) * WW + w0 + row;
        v16bf a;
#pragma unroll
        for (int e = 0; e < 16; ++e) {
            const int ko = (e < 8) ? e : e + 8;
            a[e] = (bf16)srcRow[(size_t)ko * HW];
        }
#pragma unroll
        for (int j = 0; j < 5; ++j) {
            const v16bf bfrag = *(const v16bf*)(wf + ((size_t)(j * 2 + kc) << 9) + (lane << 4));
            acc[j] = __builtin_amdgcn_wmma_f32_16x16x32_bf16(false, a, false, bfrag,
                                                             (short)0, acc[j], false, false);
        }
    }
    const int mbase = (lane & 16) ? 8 : 0;
#pragma unroll
    for (int j = 0; j < 5; ++j) {
        const int oc = j * 16 + (lane & 15);
#pragma unroll
        for (int i = 0; i < 8; ++i) {
            const size_t pix = ((size_t)b * HH + h) * WW + (w0 + mbase + i);
            const float  val = acc[j][i];
            if      (oc < 8)  q[pix * CQ + oc] = val;
            else if (oc < 16) k[pix * CQ + (oc - 8)] = val;
            else              v[pix * CI + (oc - 16)] = val;
        }
    }
}

__device__ __forceinline__ float dot8(const float4 a0, const float4 a1, const float* p) {
    const float4 b0 = *(const float4*)p;
    const float4 b1 = *(const float4*)(p + 4);
    return a0.x * b0.x + a0.y * b0.y + a0.z * b0.z + a0.w * b0.w +
           a1.x * b1.x + a1.y * b1.y + a1.z * b1.z + a1.w * b1.w;
}

// ---------------------------------------------------------------------------
// CCA stage 1: per-pixel wave computes the 256 masked logits + softmax and
// stores att as bf16 in (b, w, h, 256) order, A-fragment-friendly (16B/lane).
// ---------------------------------------------------------------------------
__global__ void __launch_bounds__(32)
cca_softmax(const float* __restrict__ q, const float* __restrict__ k,
            bf16* __restrict__ attg) {
    const int w = blockIdx.x, h = blockIdx.y, b = blockIdx.z;
    const int lane = threadIdx.x;

    const size_t pixq = (((size_t)b * HH + h) * WW + w) * CQ;
    const float4 q0 = *(const float4*)(q + pixq);
    const float4 q1 = *(const float4*)(q + pixq + 4);

    float e8[8];
    float mx = -3.0e38f;
#pragma unroll
    for (int t = 0; t < 8; ++t) {
        const int jg = lane * 8 + t;
        float eacc;
        if (jg < HH) {                    // H direction (column), mask j==h
            const int j = jg;
            if (j == h) eacc = -3.0e38f;
            else        eacc = dot8(q0, q1, k + (((size_t)b * HH + j) * WW + w) * CQ);
        } else {                          // W direction (row), unmasked
            const int j = jg - HH;
            eacc = dot8(q0, q1, k + (((size_t)b * HH + h) * WW + j) * CQ);
        }
        e8[t] = eacc;
        mx = fmaxf(mx, eacc);
    }
    for (int off = 16; off >= 1; off >>= 1) mx = fmaxf(mx, __shfl_xor(mx, off, 32));
    float ssum = 0.f;
#pragma unroll
    for (int t = 0; t < 8; ++t) { e8[t] = __expf(e8[t] - mx); ssum += e8[t]; }
    for (int off = 16; off >= 1; off >>= 1) ssum += __shfl_xor(ssum, off, 32);
    const float rs = 1.f / ssum;

    v8bf st;
#pragma unroll
    for (int t = 0; t < 8; ++t) st[t] = (bf16)(e8[t] * rs);
    *(v8bf*)(attg + (((size_t)b * WW + w) * HH + h) * 256 + lane * 8) = st;
}

// ---------------------------------------------------------------------------
// CCA stage 2: att-weighted aggregation as WMMA GEMMs.
// DIRH : per (b,w) column,  out(h,c)  = sum_j attH(h,j) * v(b,j,w,c)
// !DIRH: per (b,h) row,     out(w,c)  = sum_j attW(w,j) * v(b,h,j,c)
// Block = 8 waves (one 16-query M-tile each) x 4 c-tiles x K=128 (4 chunks).
// ---------------------------------------------------------------------------
template<bool DIRH>
__global__ void __launch_bounds__(256)
cca_av(const bf16* __restrict__ att, const float* __restrict__ v,
       float* __restrict__ outp) {
    const int fix  = blockIdx.x;             // w for DIRH, h for !DIRH
    const int b    = blockIdx.y;
    const int wave = threadIdx.x >> 5;       // M tile 0..7
    const int lane = threadIdx.x & 31;
    const int row  = lane & 15;
    const int kbA  = (lane & 16) ? 16 : 0;   // A K sub-base
    const int kbB  = (lane & 16) ? 16 : 0;   // B K sub-base
    const int n    = lane & 15;
    const int q0   = wave * 16;

    const size_t qpix = DIRH ? (((size_t)b * WW + fix) * HH + (q0 + row))
                             : (((size_t)b * WW + (q0 + row)) * HH + fix);
    const bf16* arow = att + qpix * 256 + (DIRH ? 0 : 128) + kbA;

    v8f acc[4];
#pragma unroll
    for (int ct = 0; ct < 4; ++ct) acc[ct] = (v8f){};

#pragma unroll
    for (int kc = 0; kc < 4; ++kc) {
        const v8bf lo = *(const v8bf*)(arow + kc * 32);
        const v8bf hi = *(const v8bf*)(arow + kc * 32 + 8);
        v16bf a;
#pragma unroll
        for (int e = 0; e < 8; ++e) { a[e] = lo[e]; a[e + 8] = hi[e]; }
#pragma unroll
        for (int ct = 0; ct < 4; ++ct) {
            v16bf bf_;
#pragma unroll
            for (int e = 0; e < 16; ++e) {
                const int j = kc * 32 + kbB + e;
                const size_t bpix = DIRH ? (((size_t)b * HH + j) * WW + fix)
                                         : (((size_t)b * HH + fix) * WW + j);
                bf_[e] = (bf16)v[bpix * CI + ct * 16 + n];
            }
            acc[ct] = __builtin_amdgcn_wmma_f32_16x16x32_bf16(false, a, false, bf_,
                                                              (short)0, acc[ct], false, false);
        }
    }
    const int mb = (lane & 16) ? 8 : 0;
#pragma unroll
    for (int ct = 0; ct < 4; ++ct) {
        const int c = ct * 16 + n;
#pragma unroll
        for (int i = 0; i < 8; ++i) {
            const int qq = q0 + mb + i;
            const size_t opix = DIRH ? (((size_t)b * WW + fix) * HH + qq)
                                     : (((size_t)b * HH + fix) * WW + qq);
            outp[opix * CI + c] = acc[ct][i];
        }
    }
}

// ---------------------------------------------------------------------------
// CCA stage 3: out = gamma*(outH+outW) + x, channel-last -> channel-first via
// padded LDS transpose (64px x 64ch tile), coalesced reads and writes.
// ---------------------------------------------------------------------------
__global__ void __launch_bounds__(256)
cca_combine(const float* __restrict__ oh, const float* __restrict__ ow,
            const float* __restrict__ xin, const float* __restrict__ gamma_p,
            float* __restrict__ out) {
    __shared__ float tile[64 * 65];
    const int w0 = blockIdx.x * 64;
    const int h  = blockIdx.y;
    const int b  = blockIdx.z;
    const int tid = threadIdx.x;
    const float gamma = *gamma_p;

#pragma unroll
    for (int it = 0; it < 16; ++it) {
        const int lin = it * 256 + tid;
        const int c = lin & 63;
        const int p = lin >> 6;
        const int w = w0 + p;
        const float vh = oh[(((size_t)b * WW + w) * HH + h) * CI + c];
        const float vw = ow[(((size_t)b * HH + h) * WW + w) * CI + c];
        tile[c * 65 + p] = vh + vw;
    }
    __syncthreads();
#pragma unroll
    for (int it = 0; it < 16; ++it) {
        const int lin = it * 256 + tid;
        const int p = lin & 63;
        const int c = lin >> 6;
        const size_t o = (((size_t)b * CI + c) * HH + h) * WW + w0 + p;
        out[o] = gamma * tile[c * 65 + p] + xin[o];
    }
}

// Final 1x1 projection 32->32 with bias: one wave = 16 pixels x 32 outs,
// K = 32 = one bf16 WMMA chunk, A fragment shared by both col-tiles.
__global__ void __launch_bounds__(32)
final_wmma(const float* __restrict__ x, const bf16* __restrict__ wf,
           const float* __restrict__ bias, float* __restrict__ out) {
    const int wt = blockIdx.x;                  // 0..7
    const int h  = blockIdx.y, b = blockIdx.z;
    const int lane  = threadIdx.x;
    const int w0    = wt * 16;
    const int row   = lane & 15;
    const int kbase = (lane & 16) ? 8 : 0;
    const float* srcRow = x + (((size_t)b * CO + kbase) * HH + h) * WW + w0 + row;
    v16bf a;
#pragma unroll
    for (int e = 0; e < 16; ++e) {
        const int ko = (e < 8) ? e : e + 8;
        a[e] = (bf16)srcRow[(size_t)ko * HW];
    }
    const int mbase = (lane & 16) ? 8 : 0;
#pragma unroll
    for (int j = 0; j < 2; ++j) {
        v8f acc = {};
        const v16bf bfrag = *(const v16bf*)(wf + ((size_t)j << 9) + (lane << 4));
        acc = __builtin_amdgcn_wmma_f32_16x16x32_bf16(false, a, false, bfrag,
                                                      (short)0, acc, false, false);
        const int   oc = j * 16 + (lane & 15);
        const float bv = bias[oc];
        float* op = out + (((size_t)b * CO + oc) * HH + h) * WW + w0 + mbase;
        *(float4*)op       = make_float4(acc[0] + bv, acc[1] + bv, acc[2] + bv, acc[3] + bv);
        *(float4*)(op + 4) = make_float4(acc[4] + bv, acc[5] + bv, acc[6] + bv, acc[7] + bv);
    }
}

// ---------------------------------------------------------------------------
extern "C" void kernel_launch(void* const* d_in, const int* in_sizes, int n_in,
                              void* d_out, int out_size, void* d_ws, size_t ws_size,
                              hipStream_t stream) {
    (void)in_sizes; (void)n_in; (void)out_size; (void)ws_size;
    const float* x     = (const float*)d_in[0];
    const float* Wa    = (const float*)d_in[1];
    const float* ga    = (const float*)d_in[2];
    const float* ba    = (const float*)d_in[3];
    const float* Wq    = (const float*)d_in[4];
    const float* Wk    = (const float*)d_in[5];
    const float* Wv    = (const float*)d_in[6];
    const float* gamma = (const float*)d_in[7];
    const float* Wb    = (const float*)d_in[8];
    const float* gb    = (const float*)d_in[9];
    const float* bb    = (const float*)d_in[10];
    const float* Wbt   = (const float*)d_in[11];
    const float* gbt   = (const float*)d_in[12];
    const float* bbt   = (const float*)d_in[13];
    const float* Wf    = (const float*)d_in[14];
    const float* bfb   = (const float*)d_in[15];
    // recurrence == 2 (fixed by setup; device scalar not host-readable under capture)

    char*  ws  = (char*)d_ws;
    size_t off = 0;
    auto alloc = [&](size_t bytes) -> void* {
        void* p = ws + off;
        off = (off + bytes + 255) & ~(size_t)255;
        return p;
    };
    const size_t szCI = NPIX * CI * sizeof(float);
    float* bufA   = (float*)alloc(szCI);                      // conv_a output
    float* bufP0  = (float*)alloc(szCI);                      // cca ping
    float* bufP1  = (float*)alloc(szCI);                      // cca pong
    float* bufQ   = (float*)alloc(NPIX * CQ * sizeof(float));
    float* bufK   = (float*)alloc(NPIX * CQ * sizeof(float));
    float* bufV   = (float*)alloc(szCI);
    float* bufOH  = (float*)alloc(szCI);                      // outH partial
    float* bufOW  = (float*)alloc(szCI);                      // outW partial
    bf16*  bufATT = (bf16*)alloc(NPIX * 256 * sizeof(bf16));  // softmax weights
    float* bufBT  = (float*)alloc(NPIX * CO * sizeof(float)); // conv_bt output
    bf16*  packA  = (bf16*)alloc(4 * 9 * 8 * 512 * sizeof(bf16));
    bf16*  packB  = (bf16*)alloc(4 * 9 * 2 * 512 * sizeof(bf16));
    bf16*  packBT = (bf16*)alloc(2 * 9 * 10 * 512 * sizeof(bf16));
    bf16*  packQKV= (bf16*)alloc(5 * 2 * 512 * sizeof(bf16));
    bf16*  packF  = (bf16*)alloc(2 * 1 * 512 * sizeof(bf16));
    float* stats  = (float*)alloc(320 * sizeof(float));
    float* sumA  = stats,        *sqA  = stats + 64;
    float* sumB  = stats + 128,  *sqB  = stats + 192;
    float* sumBT = stats + 256,  *sqBT = stats + 288;

    hipMemsetAsync(stats, 0, 320 * sizeof(float), stream);

    // Pack all weights to bf16 WMMA fragments
    pack_conv_weights<<<(4*9*8*512 + 255) / 256, 256, 0, stream>>>(Wa,  packA,  64, 256, 9);
    pack_conv_weights<<<(4*9*2*512 + 255) / 256, 256, 0, stream>>>(Wb,  packB,  64,  64, 9);
    pack_conv_weights<<<(2*9*10*512 + 255) / 256, 256, 0, stream>>>(Wbt, packBT, 32, 320, 9);
    pack_conv_weights<<<(2*1*512 + 255) / 256, 256, 0, stream>>>(Wf,  packF,  32,  32, 1);
    pack_qkv_weights<<<(5*2*512 + 255) / 256, 256, 0, stream>>>(Wq, Wk, Wv, packQKV);

    const dim3 gridConv64((WW/16) * 2, HH, BB);   // NT=2 -> 2 tile-groups of 32 oc
    const dim3 gridConv32((WW/16) * 1, HH, BB);   // NT=2 -> all 32 oc in one wave
    const dim3 gridW16(WW/16, HH, BB);
    const dim3 gridPix(WW, HH, BB);
    const dim3 gridAvH(WW, BB);
    const dim3 gridAvW(HH, BB);
    const dim3 gridCmb(WW/64, HH, BB);

    // conv_a + ABN
    conv3x3_wmma<2><<<gridConv64, 32, 0, stream>>>(x, 256, x, 0, packA, bufA, sumA, sqA, 64);
    abn_apply<<<(int)((NPIX*CI/4 + 255) / 256), 256, 0, stream>>>(bufA, sumA, sqA, ga, ba, 64, (int)(NPIX*CI/4));

    // CCA recurrence (2 iterations), each: qkv -> softmax -> 2x WMMA AV -> combine
    qkv_wmma<<<gridW16, 32, 0, stream>>>(bufA, packQKV, bufQ, bufK, bufV);
    cca_softmax<<<gridPix, 32, 0, stream>>>(bufQ, bufK, bufATT);
    cca_av<true ><<<gridAvH, 256, 0, stream>>>(bufATT, bufV, bufOH);
    cca_av<false><<<gridAvW, 256, 0, stream>>>(bufATT, bufV, bufOW);
    cca_combine<<<gridCmb, 256, 0, stream>>>(bufOH, bufOW, bufA, gamma, bufP0);

    qkv_wmma<<<gridW16, 32, 0, stream>>>(bufP0, packQKV, bufQ, bufK, bufV);
    cca_softmax<<<gridPix, 32, 0, stream>>>(bufQ, bufK, bufATT);
    cca_av<true ><<<gridAvH, 256, 0, stream>>>(bufATT, bufV, bufOH);
    cca_av<false><<<gridAvW, 256, 0, stream>>>(bufATT, bufV, bufOW);
    cca_combine<<<gridCmb, 256, 0, stream>>>(bufOH, bufOW, bufP0, gamma, bufP1);

    // conv_b + ABN (reuse bufP0)
    conv3x3_wmma<2><<<gridConv64, 32, 0, stream>>>(bufP1, 64, bufP1, 0, packB, bufP0, sumB, sqB, 64);
    abn_apply<<<(int)((NPIX*CI/4 + 255) / 256), 256, 0, stream>>>(bufP0, sumB, sqB, gb, bb, 64, (int)(NPIX*CI/4));

    // conv_bt on concat(x, bufP0) + ABN
    conv3x3_wmma<2><<<gridConv32, 32, 0, stream>>>(x, 256, bufP0, 64, packBT, bufBT, sumBT, sqBT, 32);
    abn_apply<<<(int)((NPIX*CO/4 + 255) / 256), 256, 0, stream>>>(bufBT, sumBT, sqBT, gbt, bbt, 32, (int)(NPIX*CO/4));

    // final 1x1 + bias -> d_out
    final_wmma<<<gridW16, 32, 0, stream>>>(bufBT, packF, bfb, (float*)d_out);
}
